// HyperProjectionHead_77962246357280
// MI455X (gfx1250) — matrix operand order repaired and checked by reading, not compiled, not run on hardware
//
#include <hip/hip_runtime.h>

// ---------------------------------------------------------------------------
// HyperProjectionHead on MI455X (gfx1250):
//   out[b,p] = sum_{k,i} h[b,k]*x[b,i]*W2[p*512+i,k] + sum_i b2[p*512+i]*x[b,i]
// One bf16 WMMA GEMM  M=8192, N=128, K=131072+512, A generated on the fly
// as the rank-1 product h[b,:] (x) x[b,:].  W2 pre-packed bf16 [k][p][i].
// B tiles staged with CDNA5 async global->LDS (ASYNCcnt) when available.
// ---------------------------------------------------------------------------

typedef __attribute__((ext_vector_type(16))) __bf16 v16bf;
typedef __attribute__((ext_vector_type(8)))  float  v8f;
typedef __attribute__((ext_vector_type(8)))  unsigned short v8us;
typedef __attribute__((ext_vector_type(4)))  int    v4i;

#define B_ROWS 8192
#define IN_DIM 512
#define HID    256
#define PROJ   128
#define KTOT   257   // 256 h-slots + 1 bias(b2) slot with h==1

#define AS1 __attribute__((address_space(1)))
#define AS3 __attribute__((address_space(3)))

#if __has_builtin(__builtin_amdgcn_global_load_async_to_lds_b128) && \
    __has_builtin(__builtin_amdgcn_s_wait_asynccnt)
#define USE_ASYNC_LDS 1
#else
#define USE_ASYNC_LDS 0
#endif

static __device__ __forceinline__ unsigned short f2bf(float f) {
  return __builtin_bit_cast(unsigned short, (__bf16)f);
}

// generic->LDS and generic->global address-space casts (via integer; the low
// 32 bits of a generic LDS address are the LDS byte offset on CDNA5)
static __device__ __forceinline__ AS3 v4i* to_lds(void* p) {
  return (AS3 v4i*)(unsigned int)(unsigned long long)p;
}
static __device__ __forceinline__ AS1 v4i* to_glb(const void* p) {
  return (AS1 v4i*)(unsigned long long)p;
}

// async 16B global->LDS copy (falls back to load+store pre-async toolchains)
static __device__ __forceinline__ void cp16_g2l(unsigned short* l,
                                                const unsigned short* g) {
#if USE_ASYNC_LDS
  __builtin_amdgcn_global_load_async_to_lds_b128(to_glb(g), to_lds(l), 0, 0);
#else
  *(uint4*)l = *(const uint4*)g;
#endif
}
static __device__ __forceinline__ void wait_async() {
#if USE_ASYNC_LDS
  __builtin_amdgcn_s_wait_asynccnt(0);
#endif
}

// ---------------------------------------------------------------------------
// Repack W2 [p*512+i, k] (f32, k contiguous) -> w2p bf16 [k][p][i] (i contiguous)
// via LDS transpose so both global read and global write are coalesced.
// grid = (128 p, 16 i-tiles, 4 k-tiles), block = 256
// ---------------------------------------------------------------------------
__global__ __launch_bounds__(256) void pack_w2_kernel(
    const float* __restrict__ W2, unsigned short* __restrict__ w2p) {
  __shared__ float tile[32][66];                    // 32 i x 64 k (+pad)
  const int t  = threadIdx.x;
  const int p  = blockIdx.x;
  const int i0 = blockIdx.y * 32;
  const int k0 = blockIdx.z * 64;
  {
    int ir = t >> 3;                                // 0..31
    int kc = (t & 7) * 8;                           // 0..56
    const float* src = W2 + ((size_t)(p * IN_DIM + i0 + ir)) * HID + k0 + kc;
    float4 a = *(const float4*)(src);
    float4 b = *(const float4*)(src + 4);
    tile[ir][kc + 0] = a.x; tile[ir][kc + 1] = a.y;
    tile[ir][kc + 2] = a.z; tile[ir][kc + 3] = a.w;
    tile[ir][kc + 4] = b.x; tile[ir][kc + 5] = b.y;
    tile[ir][kc + 6] = b.z; tile[ir][kc + 7] = b.w;
  }
  __syncthreads();
  {
    int kc = t >> 2;                                // 0..63
    int iq = (t & 3) * 8;                           // 0..24
    v8us v;
    #pragma unroll
    for (int j = 0; j < 8; ++j) v[j] = f2bf(tile[iq + j][kc]);
    unsigned short* dst = w2p + ((size_t)(k0 + kc) * PROJ + p) * IN_DIM + i0 + iq;
    *(v8us*)dst = v;
  }
}

// b2 [p*512+i] f32 -> bf16 into virtual slot k=256 of w2p (same [p][i] layout).
__global__ __launch_bounds__(256) void pack_b2_kernel(
    const float* __restrict__ b2, unsigned short* __restrict__ w2p) {
  int idx = (blockIdx.x * 256 + threadIdx.x) * 8;
  float4 a = *(const float4*)(b2 + idx);
  float4 b = *(const float4*)(b2 + idx + 4);
  v8us v;
  v[0] = f2bf(a.x); v[1] = f2bf(a.y); v[2] = f2bf(a.z); v[3] = f2bf(a.w);
  v[4] = f2bf(b.x); v[5] = f2bf(b.y); v[6] = f2bf(b.z); v[7] = f2bf(b.w);
  *(v8us*)(w2p + (size_t)256 * PROJ * IN_DIM + idx) = v;
}

// ---------------------------------------------------------------------------
// Stage 1: h = relu(fused @ W1^T + b1), bf16 out.  M=8192 N=256 K=512.
// One wave = one 16x16 tile, K-loop of 16 x (K=32) WMMAs.  8 waves/WG.
// grid = 8192 tiles / 8 = 1024 blocks.
// ---------------------------------------------------------------------------
__global__ __launch_bounds__(256) void gen_h_kernel(
    const float* __restrict__ fused, const float* __restrict__ W1,
    const float* __restrict__ b1, unsigned short* __restrict__ hbuf) {
  const int tid  = threadIdx.x;
  const int lane = tid & 31;
  const int wave = tid >> 5;
  const int tileId = blockIdx.x * 8 + wave;         // 0..8191
  const int m = tileId >> 4;                        // 0..511  (rows/16)
  const int n = tileId & 15;                        // 0..15   (cols/16)
  const int half = (lane & 16) ? 1 : 0;
  const int r = m * 16 + (lane & 15);               // A row (M = lane%16)
  const int c = n * 16 + (lane & 15);               // B col (N = lane%16)

  v8f acc = {};
  const float* arow = fused + (size_t)r * IN_DIM + half * 8;   // A K-halves 0..7 / 8..15 (+16)
  const float* brow = W1    + (size_t)c * IN_DIM + half * 8;   // B K-halves 0..7 / 8..15 (+16)

  for (int kk = 0; kk < 16; ++kk) {
    const int i0 = kk * 32;
    float4 a0 = *(const float4*)(arow + i0);
    float4 a1 = *(const float4*)(arow + i0 + 4);
    float4 a2 = *(const float4*)(arow + i0 + 16);
    float4 a3 = *(const float4*)(arow + i0 + 20);
    float4 b0 = *(const float4*)(brow + i0);
    float4 b1v = *(const float4*)(brow + i0 + 4);
    float4 b2v = *(const float4*)(brow + i0 + 16);
    float4 b3v = *(const float4*)(brow + i0 + 20);
    v16bf a, b;
    a[0]=(__bf16)a0.x;  a[1]=(__bf16)a0.y;  a[2]=(__bf16)a0.z;  a[3]=(__bf16)a0.w;
    a[4]=(__bf16)a1.x;  a[5]=(__bf16)a1.y;  a[6]=(__bf16)a1.z;  a[7]=(__bf16)a1.w;
    a[8]=(__bf16)a2.x;  a[9]=(__bf16)a2.y;  a[10]=(__bf16)a2.z; a[11]=(__bf16)a2.w;
    a[12]=(__bf16)a3.x; a[13]=(__bf16)a3.y; a[14]=(__bf16)a3.z; a[15]=(__bf16)a3.w;
    b[0]=(__bf16)b0.x;  b[1]=(__bf16)b0.y;  b[2]=(__bf16)b0.z;  b[3]=(__bf16)b0.w;
    b[4]=(__bf16)b1v.x; b[5]=(__bf16)b1v.y; b[6]=(__bf16)b1v.z; b[7]=(__bf16)b1v.w;
    b[8]=(__bf16)b2v.x; b[9]=(__bf16)b2v.y; b[10]=(__bf16)b2v.z;b[11]=(__bf16)b2v.w;
    b[12]=(__bf16)b3v.x;b[13]=(__bf16)b3v.y;b[14]=(__bf16)b3v.z;b[15]=(__bf16)b3v.w;
    acc = __builtin_amdgcn_wmma_f32_16x16x32_bf16(false, a, false, b,
                                                  (short)0, acc, false, false);
  }
  const float bias = b1[c];
  #pragma unroll
  for (int v = 0; v < 8; ++v) {                     // C: VGPR v -> M = v + half*8
    float f = acc[v] + bias;
    f = f > 0.f ? f : 0.f;
    int row = m * 16 + v + half * 8;
    ((__bf16*)hbuf)[(size_t)row * HID + c] = (__bf16)f;
  }
}

// ---------------------------------------------------------------------------
// Stage 2: the big fused GEMM.  block = 256 thr = 8 waves, BM = 128 rows.
// Wave w owns rows [blk*128 + w*16, +16) x all 128 p  => 8 v8f accum tiles.
// LDS: h tile 64KB + double-buffered 8KB B tiles (bf16 [p][i], 64B rows),
// staged via async global->LDS.  Inner loop over k (257): A-frag =
// h[b,k] * x[b, i0..i0+31] built in registers from packed cvt.
// grid = 64 blocks.
// ---------------------------------------------------------------------------
__global__ __launch_bounds__(256) void hyper_main_kernel(
    const float* __restrict__ x,
    const unsigned short* __restrict__ w2p,
    const unsigned short* __restrict__ hbuf,
    float* __restrict__ out) {
  extern __shared__ unsigned short smem[];
  unsigned short* sh = smem;                        // 128*256 bf16 = 64KB
  unsigned short* sB = smem + 128 * HID;            // 2 * 128*32 bf16 = 16KB

  const int tid  = threadIdx.x;
  const int lane = tid & 31;
  const int wave = tid >> 5;
  const int half = (lane & 16) ? 1 : 0;
  const int rowBase = blockIdx.x * 128;
  const int rloc = wave * 16 + (lane & 15);         // local row 0..127

  // preload h tile: 128 rows x 256 k (bf16), async global->LDS
  {
    const unsigned short* src = hbuf + (size_t)rowBase * HID;
    #pragma unroll
    for (int it = 0; it < 16; ++it) {
      int o = (it * 256 + tid) * 8;
      cp16_g2l(sh + o, src + o);
    }
  }

  // cooperative stage of one 32i x 128p bf16 B-tile (8KB) into LDS [p][i]
  auto stage = [&](int kb, int k, int i0) {
    unsigned short* dst = sB + kb * 4096;
    const unsigned short* src = w2p + (size_t)k * (PROJ * IN_DIM) + i0;
    int c = tid;
    #pragma unroll
    for (int it = 0; it < 2; ++it, c += 256) {      // 512 x 16B chunks
      int p = c >> 2, q = c & 3;
      cp16_g2l(dst + p * 32 + q * 8, src + (size_t)p * IN_DIM + q * 8);
    }
    // keep the L2-resident W2 stream ahead of the async engine (k+2 tile)
    __builtin_prefetch(src + 2 * (size_t)(PROJ * IN_DIM) + (size_t)(tid >> 2) * IN_DIM,
                       0, 1);
  };

  v8f acc[8] = {};                                  // 16 rows x 128 p, f32
  const float* xr_base = x + (size_t)(rowBase + rloc) * IN_DIM;

  for (int ic = 0; ic < 16; ++ic) {                 // i chunks of 32
    const int i0 = ic * 32;
    // x fragment for this wave's 16 rows, held in regs across the whole k loop
    const float* xr = xr_base + i0 + half * 16;
    float4 xa0 = *(const float4*)(xr);
    float4 xa1 = *(const float4*)(xr + 4);
    float4 xa2 = *(const float4*)(xr + 8);
    float4 xa3 = *(const float4*)(xr + 12);

    wait_async();
    __syncthreads();                                // buffers free from prev ic
    stage(0, 0, i0);
    for (int k = 0; k < KTOT; ++k) {
      wait_async();                                 // our async stores landed
      __syncthreads();                              // everyone's tile visible
      if (k + 1 < KTOT) stage((k + 1) & 1, k + 1, i0);

      float hv = (k < 256)
          ? (float)(*(const __bf16*)(sh + rloc * HID + k))
          : 1.0f;                                   // k==256: bias(b2) slot
      v16bf a;
      a[0]=(__bf16)(hv*xa0.x);  a[1]=(__bf16)(hv*xa0.y);
      a[2]=(__bf16)(hv*xa0.z);  a[3]=(__bf16)(hv*xa0.w);
      a[4]=(__bf16)(hv*xa1.x);  a[5]=(__bf16)(hv*xa1.y);
      a[6]=(__bf16)(hv*xa1.z);  a[7]=(__bf16)(hv*xa1.w);
      a[8]=(__bf16)(hv*xa2.x);  a[9]=(__bf16)(hv*xa2.y);
      a[10]=(__bf16)(hv*xa2.z); a[11]=(__bf16)(hv*xa2.w);
      a[12]=(__bf16)(hv*xa3.x); a[13]=(__bf16)(hv*xa3.y);
      a[14]=(__bf16)(hv*xa3.z); a[15]=(__bf16)(hv*xa3.w);

      const unsigned short* sb = sB + (k & 1) * 4096;
      #pragma unroll
      for (int pt = 0; pt < 8; ++pt) {
        v16bf b = *(const v16bf*)(sb + (pt * 16 + (lane & 15)) * 32 + half * 16);
        acc[pt] = __builtin_amdgcn_wmma_f32_16x16x32_bf16(false, a, false, b,
                                                          (short)0, acc[pt],
                                                          false, false);
      }
    }
  }

  // epilogue: C layout (VGPR v -> M = v + half*8, N = lane%16), f32 out
  #pragma unroll
  for (int pt = 0; pt < 8; ++pt) {
    #pragma unroll
    for (int v = 0; v < 8; ++v) {
      int row = rowBase + wave * 16 + v + half * 8;
      int col = pt * 16 + (lane & 15);
      out[(size_t)row * PROJ + col] = acc[pt][v];
    }
  }
}

// ---------------------------------------------------------------------------
extern "C" void kernel_launch(void* const* d_in, const int* in_sizes, int n_in,
                              void* d_out, int out_size, void* d_ws, size_t ws_size,
                              hipStream_t stream) {
  const float* fused = (const float*)d_in[0];
  const float* x     = (const float*)d_in[1];
  const float* W1    = (const float*)d_in[2];
  const float* b1    = (const float*)d_in[3];
  const float* W2    = (const float*)d_in[4];
  const float* b2    = (const float*)d_in[5];
  float* out = (float*)d_out;

  // workspace: w2p bf16 [257][128][512] (33.7MB) then h bf16 [8192][256] (4.2MB)
  unsigned short* w2p  = (unsigned short*)d_ws;
  unsigned short* hbuf = (unsigned short*)((char*)d_ws +
                          (size_t)KTOT * PROJ * IN_DIM * sizeof(unsigned short));

  hipLaunchKernelGGL(pack_w2_kernel, dim3(128, 16, 4), dim3(256), 0, stream, W2, w2p);
  hipLaunchKernelGGL(pack_b2_kernel, dim3(32), dim3(256), 0, stream, b2, w2p);
  hipLaunchKernelGGL(gen_h_kernel, dim3(1024), dim3(256), 0, stream,
                     fused, W1, b1, hbuf);
  size_t shmem = (size_t)(128 * HID + 2 * 128 * 32) * sizeof(unsigned short); // 80KB
  hipLaunchKernelGGL(hyper_main_kernel, dim3(B_ROWS / 128), dim3(256), shmem, stream,
                     x, w2p, hbuf, out);
}
